// ANSGN_90417651515758
// MI455X (gfx1250) — matrix operand order repaired and checked by reading, not compiled, not run on hardware
//
#include <hip/hip_runtime.h>

#define HID 64

typedef __attribute__((ext_vector_type(16))) __bf16 v16bf;
typedef __attribute__((ext_vector_type(8)))  float  v8f;

// ---------- bf16 split helpers (round-to-nearest-even) ----------
__device__ __forceinline__ unsigned short f32_bf16_rne(float x) {
    unsigned u = __builtin_bit_cast(unsigned, x);
    unsigned r = u + 0x7FFFu + ((u >> 16) & 1u);
    return (unsigned short)(r >> 16);
}
__device__ __forceinline__ float bf16b_f32(unsigned short b) {
    unsigned u = ((unsigned)b) << 16;
    return __builtin_bit_cast(float, u);
}
__device__ __forceinline__ __bf16 bf16_from_bits(unsigned short b) {
    return __builtin_bit_cast(__bf16, b);
}

// ---------- generic zero fill ----------
__global__ void ansgn_zero_f32(float* __restrict__ p, int n) {
    int i = blockIdx.x * blockDim.x + threadIdx.x;
    if (i < n) p[i] = 0.0f;
}

// ---------- degree: deg[dst] += 1 ----------
__global__ void ansgn_deg(const long long* __restrict__ dst, float* __restrict__ deg, int E) {
    int e = blockIdx.x * blockDim.x + threadIdx.x;
    if (e < E) atomicAdd(&deg[(int)dst[e]], 1.0f);
}

// ---------- dinv = rsqrt(deg + 1) in place ----------
__global__ void ansgn_dinv(float* __restrict__ d, int N) {
    int i = blockIdx.x * blockDim.x + threadIdx.x;
    if (i < N) d[i] = rsqrtf(d[i] + 1.0f);
}

// ---------- conv1 scalar scatter: spre[dst] += dinv[src]*x[src] ----------
__global__ void ansgn_spre(const long long* __restrict__ src, const long long* __restrict__ dst,
                           const float* __restrict__ dinv, const float* __restrict__ x,
                           float* __restrict__ spre, int E) {
    int e = blockIdx.x * blockDim.x + threadIdx.x;
    if (e < E) {
        int s = (int)src[e];
        atomicAdd(&spre[(int)dst[e]], dinv[s] * x[s]);
    }
}

// ---------- h1[i,f] = relu( dinv_i*(spre_i + dinv_i*x_i)*W1[f] + b1[f] ) ----------
__global__ void ansgn_h1(const float* __restrict__ dinv, const float* __restrict__ spre,
                         const float* __restrict__ x, const float* __restrict__ W1,
                         const float* __restrict__ b1, float* __restrict__ h1, int N) {
    int tid = blockIdx.x * blockDim.x + threadIdx.x;
    if (tid >= N * HID) return;
    int i = tid >> 6, f = tid & 63;
    float di = dinv[i];
    float scal = di * (spre[i] + di * x[i]);
    float v = scal * W1[f] + b1[f];
    h1[tid] = v > 0.0f ? v : 0.0f;
}

// ---------- prep: split W2 to bf16 hi/lo pack; build fused proj matrix M + bias ----------
__global__ void ansgn_prep(const float* __restrict__ W2, const float* __restrict__ inw,
                           const float* __restrict__ inb, const float* __restrict__ outw,
                           const float* __restrict__ outb,
                           unsigned* __restrict__ W2pack, unsigned* __restrict__ Mpack,
                           float* __restrict__ bcomb) {
    int idx = blockIdx.x * blockDim.x + threadIdx.x;   // (k,n)
    if (idx >= HID * HID) return;
    int k = idx >> 6, n = idx & 63;

    float w = W2[k * HID + n];
    unsigned short hb = f32_bf16_rne(w);
    unsigned short lb = f32_bf16_rne(w - bf16b_f32(hb));
    W2pack[idx] = (unsigned)hb | ((unsigned)lb << 16);

    // M[k,n] = sum_f Wv[f,k]*Wo[n,f] ; Wv row f is in_proj_w row (2*HID+f)
    float acc = 0.0f;
    for (int f = 0; f < HID; ++f)
        acc += inw[(size_t)(2 * HID + f) * HID + k] * outw[n * HID + f];
    hb = f32_bf16_rne(acc);
    lb = f32_bf16_rne(acc - bf16b_f32(hb));
    Mpack[idx] = (unsigned)hb | ((unsigned)lb << 16);

    if (k == 0) {
        float bacc = outb[n];
        for (int f = 0; f < HID; ++f) bacc += inb[2 * HID + f] * outw[n * HID + f];
        bcomb[n] = bacc;
    }
}

// ---------- WMMA GEMM: C[N,64] = A[N,64] @ B[64,64] (+bias), bf16x3 emulated fp32 ----------
// one wave = one 16x16 tile; 4 waves/block cover the 64 output columns.
__global__ __launch_bounds__(128) void ansgn_gemm64(
    const float* __restrict__ A, const unsigned* __restrict__ Bpack,
    const float* __restrict__ bias, float* __restrict__ C, int nrows, int hasBias) {
    const int lane = threadIdx.x & 31;
    const int wave = threadIdx.x >> 5;
    const int half = lane >> 4;
    const int r    = lane & 15;
    const int m0   = blockIdx.x * 16;
    const int n0   = wave * 16;

    int m  = m0 + r;                      // A row this lane supplies
    int mc = m < nrows ? m : nrows - 1;   // clamp (keeps EXEC full for WMMA)
    const float* arow = A + (size_t)mc * HID;
    const int n = n0 + r;                 // B/C column for this lane

    v8f c = {};

    #pragma unroll
    for (int kc = 0; kc < HID; kc += 32) {
        // ---- A: two contiguous 8-float runs per lane (ISA 16-bit A layout) ----
        float av[16];
        {
            const float4* p0 = (const float4*)(arow + kc + half * 8);
            const float4* p1 = (const float4*)(arow + kc + 16 + half * 8);
            float4 q0 = p0[0], q1 = p0[1], q2 = p1[0], q3 = p1[1];
            av[0]=q0.x;  av[1]=q0.y;  av[2]=q0.z;  av[3]=q0.w;
            av[4]=q1.x;  av[5]=q1.y;  av[6]=q1.z;  av[7]=q1.w;
            av[8]=q2.x;  av[9]=q2.y;  av[10]=q2.z; av[11]=q2.w;
            av[12]=q3.x; av[13]=q3.y; av[14]=q3.z; av[15]=q3.w;
        }
        v16bf ahi, alo;
        #pragma unroll
        for (int e = 0; e < 16; ++e) {
            unsigned short hb = f32_bf16_rne(av[e]);
            unsigned short lb = f32_bf16_rne(av[e] - bf16b_f32(hb));
            ahi[e] = bf16_from_bits(hb);
            alo[e] = bf16_from_bits(lb);
        }
        // ---- B: element e -> K = (e&7) + 8*half + 16*(e>>3) (+kc), column n ----
        v16bf bhi, blo;
        {
            int kb = kc + 8 * half;
            #pragma unroll
            for (int e = 0; e < 16; ++e) {
                unsigned pk = Bpack[(kb + (e & 7) + ((e >> 3) << 4)) * HID + n];
                bhi[e] = bf16_from_bits((unsigned short)(pk & 0xFFFFu));
                blo[e] = bf16_from_bits((unsigned short)(pk >> 16));
            }
        }
        c = __builtin_amdgcn_wmma_f32_16x16x32_bf16(false, ahi, false, bhi, (short)0, c, false, false);
        c = __builtin_amdgcn_wmma_f32_16x16x32_bf16(false, ahi, false, blo, (short)0, c, false, false);
        c = __builtin_amdgcn_wmma_f32_16x16x32_bf16(false, alo, false, bhi, (short)0, c, false, false);
    }

    float badd = hasBias ? bias[n] : 0.0f;
    #pragma unroll
    for (int i = 0; i < 8; ++i) {
        int row = m0 + i + 8 * half;      // C layout: vgpr i -> M = i + 8*half
        if (row < nrows) C[(size_t)row * HID + n] = c[i] + badd;
    }
}

// ---------- conv2 edge scatter: agg[dst,f] += dinv[s]*dinv[d]*t[src,f] ----------
__global__ void ansgn_scatter(const long long* __restrict__ src, const long long* __restrict__ dst,
                              const float* __restrict__ dinv, const float* __restrict__ t,
                              float* __restrict__ agg, int E) {
    long long tid = (long long)blockIdx.x * blockDim.x + threadIdx.x;
    if (tid >= (long long)E * HID) return;
    int e = (int)(tid >> 6), f = (int)(tid & 63);
    int s = (int)src[e], d = (int)dst[e];
    float norm = dinv[s] * dinv[d];
    atomicAdd(&agg[(size_t)d * HID + f], t[(size_t)s * HID + f] * norm);
}

// ---------- h2 = relu(agg + dinv^2 * t + b2), in place into agg ----------
__global__ void ansgn_h2(float* __restrict__ agg, const float* __restrict__ t,
                         const float* __restrict__ dinv, const float* __restrict__ b2, int N) {
    int tid = blockIdx.x * blockDim.x + threadIdx.x;
    if (tid >= N * HID) return;
    int i = tid >> 6, f = tid & 63;
    float di = dinv[i];
    float v = agg[tid] + di * di * t[tid] + b2[f];
    agg[tid] = v > 0.0f ? v : 0.0f;
}

// ---------- segment pool: pooled[g,f] += hf[i,f]; cnt[g] += 1 (f==0) ----------
__global__ void ansgn_pool(const long long* __restrict__ batch, const float* __restrict__ hf,
                           float* __restrict__ pooled, float* __restrict__ cnt, int N) {
    int tid = blockIdx.x * blockDim.x + threadIdx.x;
    if (tid >= N * HID) return;
    int i = tid >> 6, f = tid & 63;
    int g = (int)batch[i];
    atomicAdd(&pooled[(size_t)g * HID + f], hf[tid]);
    if (f == 0) atomicAdd(&cnt[g], 1.0f);
}

// ---------- final fc: out[g,o] = (pooled[g,:]/max(cnt,1)) . fc_w[o,:] + fc_b[o] ----------
__global__ void ansgn_fc(const float* __restrict__ pooled, const float* __restrict__ cnt,
                         const float* __restrict__ fcw, const float* __restrict__ fcb,
                         float* __restrict__ out, int G, int OUT) {
    int tid = blockIdx.x * blockDim.x + threadIdx.x;
    if (tid >= G * OUT) return;
    int g = tid / OUT, o = tid - g * OUT;
    float inv = 1.0f / fmaxf(cnt[g], 1.0f);
    float acc = 0.0f;
    for (int f = 0; f < HID; ++f) acc += pooled[(size_t)g * HID + f] * fcw[o * HID + f];
    out[tid] = acc * inv + fcb[o];
}

extern "C" void kernel_launch(void* const* d_in, const int* in_sizes, int n_in,
                              void* d_out, int out_size, void* d_ws, size_t ws_size,
                              hipStream_t stream) {
    const float*     x     = (const float*)d_in[0];
    const long long* ei    = (const long long*)d_in[1];
    const long long* batch = (const long long*)d_in[2];
    const float*     W1    = (const float*)d_in[3];
    const float*     b1    = (const float*)d_in[4];
    const float*     W2    = (const float*)d_in[5];
    const float*     b2    = (const float*)d_in[6];
    const float*     inw   = (const float*)d_in[7];
    const float*     inb   = (const float*)d_in[8];
    const float*     outw  = (const float*)d_in[9];
    const float*     outb  = (const float*)d_in[10];
    const float*     fcw   = (const float*)d_in[11];
    const float*     fcb   = (const float*)d_in[12];
    float*           out   = (float*)d_out;

    const int N   = in_sizes[0];           // x is [N,1]
    const int E   = in_sizes[1] / 2;       // edge_index [2,E]
    const int OUT = in_sizes[12];          // fc_b size
    const int G   = out_size / OUT;

    const long long* esrc = ei;
    const long long* edst = ei + E;

    // ---- workspace layout (4-byte units) ----
    float*  wsf    = (float*)d_ws;
    size_t  nA     = (size_t)N * HID;
    float*  bufA   = wsf;                        // h1 -> agg2 -> h2
    float*  bufB   = wsf + nA;                   // t  -> hf
    float*  dinv   = wsf + 2 * nA;               // deg then dinv, [N]
    float*  spre   = dinv + N;                   // [N]
    float*  pooled = spre + N;                   // [G*HID] (+cnt right after)
    float*  cnt    = pooled + (size_t)G * HID;   // [G]
    unsigned* W2pack = (unsigned*)(cnt + G);     // [HID*HID]
    unsigned* Mpack  = W2pack + HID * HID;       // [HID*HID]
    float*  bcomb  = (float*)(Mpack + HID * HID);// [HID]

    const int TB = 256;
    auto blocks = [](long long nthreads, int tb) { return (int)((nthreads + tb - 1) / tb); };

    // zero accumulators (every call; pooled and cnt are contiguous)
    ansgn_zero_f32<<<blocks(N, TB), TB, 0, stream>>>(dinv, N);
    ansgn_zero_f32<<<blocks(N, TB), TB, 0, stream>>>(spre, N);
    ansgn_zero_f32<<<blocks((long long)G * HID + G, TB), TB, 0, stream>>>(pooled, G * HID + G);

    // degree + dinv
    ansgn_deg<<<blocks(E, TB), TB, 0, stream>>>(edst, dinv, E);
    ansgn_dinv<<<blocks(N, TB), TB, 0, stream>>>(dinv, N);

    // conv1 (scalar feature path): scalar scatter, then expand through W1 + relu
    ansgn_spre<<<blocks(E, TB), TB, 0, stream>>>(esrc, edst, dinv, x, spre, E);
    ansgn_h1<<<blocks((long long)N * HID, TB), TB, 0, stream>>>(dinv, spre, x, W1, b1, bufA, N);

    // prep matrices (W2 bf16-split; fused attention projection M and bias)
    ansgn_prep<<<blocks(HID * HID, TB), TB, 0, stream>>>(W2, inw, inb, outw, outb,
                                                         W2pack, Mpack, bcomb);

    // t = h1 @ W2   (WMMA, bf16x3)
    ansgn_gemm64<<<(N + 15) / 16, 128, 0, stream>>>(bufA, W2pack, nullptr, bufB, N, 0);

    // conv2 aggregation over edges (+ self loop, bias, relu)
    ansgn_zero_f32<<<blocks((long long)nA, TB), TB, 0, stream>>>(bufA, (int)nA);
    ansgn_scatter<<<blocks((long long)E * HID, TB), TB, 0, stream>>>(esrc, edst, dinv, bufB, bufA, E);
    ansgn_h2<<<blocks((long long)N * HID, TB), TB, 0, stream>>>(bufA, bufB, dinv, b2, N);

    // hf = h2 @ M + bcomb   (fused V-proj + out-proj, WMMA, bf16x3)
    ansgn_gemm64<<<(N + 15) / 16, 128, 0, stream>>>(bufA, Mpack, bcomb, bufB, N, 1);

    // mean pool per graph, then final fc
    ansgn_pool<<<blocks((long long)N * HID, TB), TB, 0, stream>>>(batch, bufB, pooled, cnt, N);
    ansgn_fc<<<blocks((long long)G * OUT, TB), TB, 0, stream>>>(pooled, cnt, fcw, fcb, out, G, OUT);
}
